// Rel_Attention_50242527428632
// MI455X (gfx1250) — matrix operand order repaired and checked by reading, not compiled, not run on hardware
//
#include <hip/hip_runtime.h>
#include <hip/hip_bf16.h>

// ---------------------------------------------------------------------------
// Swin-style window attention for MI455X (gfx1250), all matmuls via
// v_wmma_f32_16x16x32_f16 (wave32, f16 in / f32 accumulate).
//
// Shapes: BW=2048 windows, C=256, HEADS=8, HD=32, N=49 (padded to 64).
// LDS row strides are padded (264/72 halfs, 65 floats) so per-lane b128
// fragment reads are bank-conflict-free; GEMMs run 2 M-tiles per iteration
// with independent accumulators so A-loads overlap WMMA issue.
// ---------------------------------------------------------------------------

typedef _Float16 half_t;
typedef half_t v16h __attribute__((ext_vector_type(16)));
typedef half_t v8h  __attribute__((ext_vector_type(8)));
typedef float  v8f  __attribute__((ext_vector_type(8)));

#define DEVI __device__ __forceinline__

static constexpr int BW    = 2048;   // 32*8*8 windows
static constexpr int C     = 256;
static constexpr int HEADS = 8;
static constexpr int HD    = 32;
static constexpr int NTOK  = 49;
static constexpr int NP    = 64;     // padded token count
static constexpr int K3    = 768;    // 3*C
static constexpr float SCALE = 0.17677669529663687f; // 32^-0.5

// padded LDS row strides (in elements) to kill bank conflicts
static constexpr int LDX = 264;      // XhT / OcT rows of 256 halfs (+8 pad)
static constexpr int LDP = 72;       // P / Vt rows of 64 halfs (+8 pad)
static constexpr int LDSS = 65;      // S rows of 64 floats (+1 pad)

DEVI v8f wmma_f16(v16h a, v16h b, v8f c) {
  // emits v_wmma_f32_16x16x32_f16
  return __builtin_amdgcn_wmma_f32_16x16x32_f16(
      /*neg_a=*/false, a, /*neg_b=*/false, b,
      /*c_mod=*/(short)0, c, /*reuse_a=*/false, /*reuse_b=*/false);
}

// Build a fragment from two 16-byte chunks (works for both A and B layouts):
//  A (16x32): lane r holds row M=r&15, K(e) = (e&8?16:0)+hi*8+(e&7)
//             -> chunks at column offsets hi*8 and 16+hi*8 (p1 = p0+16)
//  B (32x16): lane holds column N=lane&15, same K(e) as A -> chunks at
//             k offsets hi*8 and 16+hi*8 (p1 = p0+16)
DEVI v16h frag_ld(const half_t* p0, const half_t* p1) {
  v8h x0 = *(const v8h*)p0;
  v8h x1 = *(const v8h*)p1;
  v16h a;
#pragma unroll
  for (int e = 0; e < 8; ++e) { a[e] = x0[e]; a[8 + e] = x1[e]; }
  return a;
}

DEVI v16h zero16() {
  v16h a;
#pragma unroll
  for (int e = 0; e < 16; ++e) a[e] = (half_t)0.f;
  return a;
}

// ---------------------------------------------------------------------------
// Kernel 0: convert qkv_w [768*256] and proj_w [256*256] f32 -> f16
// ---------------------------------------------------------------------------
__global__ __launch_bounds__(256)
void convert_weights(const float* __restrict__ qkv_w,
                     const float* __restrict__ proj_w,
                     half_t* __restrict__ Wq, half_t* __restrict__ Wp) {
  int i = blockIdx.x * 256 + threadIdx.x;
  const int nq = K3 * C;           // 196608
  const int np = C * C;            // 65536
  if (i < nq) Wq[i] = (half_t)qkv_w[i];
  else if (i < nq + np) Wp[i - nq] = (half_t)proj_w[i - nq];
}

// ---------------------------------------------------------------------------
// Kernel 1: per-window QKV GEMM:  qkv[o][n] = sum_c Wq[o][c] * x[c][n] + b[o]
// Writes q (pre-scaled), k, v as f16 in [bw][h][n][d] layout.
// Block = 1 window, 256 threads (8 waves).
// ---------------------------------------------------------------------------
__global__ __launch_bounds__(256)
void qkv_kernel(const float* __restrict__ x,
                const half_t* __restrict__ Wq,
                const float* __restrict__ qkv_b,
                half_t* __restrict__ Q, half_t* __restrict__ Kg,
                half_t* __restrict__ V) {
  __shared__ half_t XhT[NP * LDX];   // B-matrix staging: [n][c], ~33 KB
  const int bw = blockIdx.x;
  const int t  = threadIdx.x;

  // Stage x[c][n] -> XhT[n][c] (f16), zero pad rows n in [49,64).
  // Lanes read consecutive n within a row of x -> coalesced.
#pragma unroll 4
  for (int i = 0; i < 64; ++i) {
    const int idx = i * 256 + t;
    const int c = idx >> 6;
    const int n = idx & 63;
    XhT[n * LDX + c] = (n < NTOK) ? (half_t)x[(bw * C + c) * NTOK + n]
                                  : (half_t)0.f;
  }
  __syncthreads();

  const int wave = t >> 5, lane = t & 31, r = lane & 15, hi = lane >> 4;
  const int tn = wave & 3;           // fixed N-tile per wave
  const int tmBase = wave >> 2;      // 0 or 1; tm = tmBase + 2*i

  // Hoist the 8 B-fragments (depend only on tn, k) into registers.
  v16h bfr[8];
#pragma unroll
  for (int k = 0; k < 8; ++k) {
    const half_t* bp = XhT + (tn * 16 + r) * LDX + k * 32 + hi * 8;
    bfr[k] = frag_ld(bp, bp + 16);
  }

  const int n = tn * 16 + r;
  const int nok = (n < NTOK);

  // 24 M-tiles per wave, processed in pairs (2 independent accumulators).
  for (int i = 0; i < 24; i += 2) {
    const int tm0 = tmBase + 2 * i;
    const int tm1 = tm0 + 2;
    const half_t* arow0 = Wq + (tm0 * 16 + r) * C + hi * 8;
    const half_t* arow1 = Wq + (tm1 * 16 + r) * C + hi * 8;
    v8f acc0 = {}, acc1 = {};
#pragma unroll
    for (int k = 0; k < 8; ++k) {
      v16h a0 = frag_ld(arow0 + k * 32, arow0 + k * 32 + 16);
      v16h a1 = frag_ld(arow1 + k * 32, arow1 + k * 32 + 16);
      acc0 = wmma_f16(a0, bfr[k], acc0);
      acc1 = wmma_f16(a1, bfr[k], acc1);
    }
    if (nok) {
#pragma unroll
      for (int u = 0; u < 2; ++u) {
        const int tm = u ? tm1 : tm0;
        const v8f acc = u ? acc1 : acc0;
        // tile-constant decode: o = tm*16 + hi*8 + j
        const int which = tm >> 4;               // 0=q 1=k 2=v
        const int h     = (tm >> 1) & 7;
        const int dbase = (tm & 1) * 16 + hi * 8;
        const float sc  = (which == 0) ? SCALE : 1.f;
        half_t* dst = ((which == 0) ? Q : (which == 1) ? Kg : V)
                      + ((bw * HEADS + h) * NTOK + n) * HD + dbase;
        const float* bb = qkv_b + tm * 16 + hi * 8;
#pragma unroll
        for (int j = 0; j < 8; ++j)
          dst[j] = (half_t)((acc[j] + bb[j]) * sc);
      }
    }
  }
}

// ---------------------------------------------------------------------------
// Kernel 2: per-window attention (all 8 heads) + fused projection.
// Block = 1 window, 256 threads (8 waves). LDS ~63 KB.
// ---------------------------------------------------------------------------
__global__ __launch_bounds__(256)
void attn_proj_kernel(const half_t* __restrict__ Q,
                      const half_t* __restrict__ Kg,
                      const half_t* __restrict__ V,
                      const float* __restrict__ bias_table,
                      const int*   __restrict__ rel_index,
                      const half_t* __restrict__ Wp,
                      const float* __restrict__ proj_b,
                      float* __restrict__ out) {
  __shared__ float  S[NP * LDSS];    // logits / exp scratch, 16.6 KB
  __shared__ half_t P[NP * LDP];     // softmax probs (f16, zero-padded), 9 KB
  __shared__ half_t Vt[HD * LDP];    // V transposed: [d][m], 4.6 KB
  __shared__ half_t OcT[NP * LDX];   // attention out: [n][c=h*32+d], ~33 KB

  const int bw = blockIdx.x, t = threadIdx.x;
  const int wave = t >> 5, lane = t & 31, r = lane & 15, hi = lane >> 4;

  for (int h = 0; h < HEADS; ++h) {
    __syncthreads();   // previous head's P/Vt/S reads complete

    // ---- stage Vt[d][m] = V[bw][h][m][d], pad m>=49 with zeros
    const half_t* Vh = V + (bw * HEADS + h) * NTOK * HD;
#pragma unroll
    for (int idx = t; idx < HD * NP; idx += 256) {
      const int d = idx >> 6, m = idx & 63;
      Vt[d * LDP + m] = (m < NTOK) ? Vh[m * HD + d] : (half_t)0.f;
    }

    // ---- S = q @ k^T : 4x4 tiles of 16x16, single K-step (hd == 32).
    // Two tiles per wave, independent accumulators.
    const half_t* Qh = Q  + (bw * HEADS + h) * NTOK * HD;
    const half_t* Kh = Kg + (bw * HEADS + h) * NTOK * HD;
#pragma unroll
    for (int tile = wave; tile < 16; tile += 8) {
      const int tm = tile >> 2, tn = tile & 3;
      const int n = tm * 16 + r;
      v16h a;
      if (n < NTOK) { const half_t* ap = Qh + n * HD + hi * 8; a = frag_ld(ap, ap + 16); }
      else          a = zero16();
      const int m = tn * 16 + r;
      v16h b;
      if (m < NTOK) { const half_t* bp = Kh + m * HD + hi * 8; b = frag_ld(bp, bp + 16); }
      else          b = zero16();
      v8f acc = {};
      acc = wmma_f16(a, b, acc);
#pragma unroll
      for (int j = 0; j < 8; ++j)
        S[(tm * 16 + hi * 8 + j) * LDSS + tn * 16 + r] = acc[j];
    }
    __syncthreads();

    // ---- zero pad region of P (so pad K-columns contribute 0 to P@V)
    for (int idx = t; idx < NP * NP; idx += 256) {
      const int n = idx >> 6, m = idx & 63;
      if (n >= NTOK || m >= NTOK) P[n * LDP + m] = (half_t)0.f;
    }

    // ---- row softmax with gathered relative position bias
    if (t < NTOK) {
      const int n = t;
      float mx = -1e30f;
      for (int m = 0; m < NTOK; ++m) {
        const float bias = bias_table[rel_index[n * NTOK + m] * HEADS + h];
        const float sv = S[n * LDSS + m] + bias;
        S[n * LDSS + m] = sv;
        mx = fmaxf(mx, sv);
      }
      float sum = 0.f;
      for (int m = 0; m < NTOK; ++m) {
        const float e = __expf(S[n * LDSS + m] - mx);
        S[n * LDSS + m] = e;
        sum += e;
      }
      const float inv = 1.f / sum;
      for (int m = 0; m < NTOK; ++m)
        P[n * LDP + m] = (half_t)(S[n * LDSS + m] * inv);
    }
    __syncthreads();

    // ---- out[n][d] = P @ V : 4x2 tiles, 2 K-steps of 32; one tile per wave
    {
      const int tm = wave >> 1, tn = wave & 1;
      v8f acc = {};
#pragma unroll
      for (int kk = 0; kk < 2; ++kk) {
        const half_t* ap = P + (tm * 16 + r) * LDP + kk * 32 + hi * 8;
        v16h a = frag_ld(ap, ap + 16);
        const half_t* bp = Vt + (tn * 16 + r) * LDP + kk * 32 + hi * 8;
        v16h b = frag_ld(bp, bp + 16);
        acc = wmma_f16(a, b, acc);
      }
#pragma unroll
      for (int j = 0; j < 8; ++j) {
        const int n = tm * 16 + hi * 8 + j;
        const int d = tn * 16 + r;
        OcT[n * LDX + h * HD + d] = (half_t)acc[j];
      }
    }
  }
  __syncthreads();   // OcT complete for all heads

  // ---- projection: y[o][n] = sum_c Wp[o][c] * OcT[n][c] + proj_b[o]
  // 16 M-tiles x 4 N-tiles = 64 tiles; 8 K-steps of 32.
  const int tn = wave & 3;
  const int tmBase = wave >> 2;

  // Hoist B-fragments (depend only on tn, k).
  v16h bfr[8];
#pragma unroll
  for (int k = 0; k < 8; ++k) {
    const half_t* bp = OcT + (tn * 16 + r) * LDX + k * 32 + hi * 8;
    bfr[k] = frag_ld(bp, bp + 16);
  }

  const int n = tn * 16 + r;
  const int nok = (n < NTOK);

  for (int i = 0; i < 8; i += 2) {
    const int tm0 = tmBase + 2 * i;
    const int tm1 = tm0 + 2;
    const half_t* arow0 = Wp + (tm0 * 16 + r) * C + hi * 8;
    const half_t* arow1 = Wp + (tm1 * 16 + r) * C + hi * 8;
    v8f acc0 = {}, acc1 = {};
#pragma unroll
    for (int k = 0; k < 8; ++k) {
      v16h a0 = frag_ld(arow0 + k * 32, arow0 + k * 32 + 16);
      v16h a1 = frag_ld(arow1 + k * 32, arow1 + k * 32 + 16);
      acc0 = wmma_f16(a0, bfr[k], acc0);
      acc1 = wmma_f16(a1, bfr[k], acc1);
    }
    if (nok) {
#pragma unroll
      for (int u = 0; u < 2; ++u) {
        const int tm = u ? tm1 : tm0;
        const v8f acc = u ? acc1 : acc0;
        const int obase = tm * 16 + hi * 8;
        float* dst = out + (bw * C + obase) * NTOK + n;
        const float* pb = proj_b + obase;
#pragma unroll
        for (int j = 0; j < 8; ++j)
          dst[j * NTOK] = acc[j] + pb[j];
      }
    }
  }
}

// ---------------------------------------------------------------------------
// Host-side launcher
// ---------------------------------------------------------------------------
extern "C" void kernel_launch(void* const* d_in, const int* in_sizes, int n_in,
                              void* d_out, int out_size, void* d_ws, size_t ws_size,
                              hipStream_t stream) {
  const float* x          = (const float*)d_in[0];
  const float* qkv_w      = (const float*)d_in[1];
  const float* qkv_b      = (const float*)d_in[2];
  const float* proj_w     = (const float*)d_in[3];
  const float* proj_b     = (const float*)d_in[4];
  const float* bias_table = (const float*)d_in[5];
  const int*   rel_index  = (const int*)d_in[6];
  float* out = (float*)d_out;

  // Workspace layout (f16), ~148 MB total:
  //   Wq [768*256] | Wp [256*256] | Q,K,V each [2048*8*49*32]
  char* p = (char*)d_ws;
  half_t* Wq = (half_t*)p;                p += (size_t)K3 * C * sizeof(half_t);
  half_t* Wp = (half_t*)p;                p += (size_t)C * C * sizeof(half_t);
  const size_t qkv_elems = (size_t)BW * HEADS * NTOK * HD;
  half_t* Q  = (half_t*)p;                p += qkv_elems * sizeof(half_t);
  half_t* Kg = (half_t*)p;                p += qkv_elems * sizeof(half_t);
  half_t* V  = (half_t*)p;                p += qkv_elems * sizeof(half_t);
  (void)ws_size; (void)in_sizes; (void)n_in; (void)out_size;

  convert_weights<<<(K3 * C + C * C + 255) / 256, 256, 0, stream>>>(qkv_w, proj_w, Wq, Wp);
  qkv_kernel<<<BW, 256, 0, stream>>>(x, Wq, qkv_b, Q, Kg, V);
  attn_proj_kernel<<<BW, 256, 0, stream>>>(Q, Kg, V, bias_table, rel_index, Wp, proj_b, out);
}
